// BottleneckFusion_73237782331601
// MI455X (gfx1250) — matrix operand order repaired and checked by reading, not compiled, not run on hardware
//
#include <hip/hip_runtime.h>
#include <hip/hip_bf16.h>

// ---------------------------------------------------------------------------
// CDNA5 (gfx1250, wave32) implementation.
//  - heavy GEMMs/convs: v_wmma_f32_16x16x32_bf16, 32Mx64N register tiles
//    (8 wmma per barrier pair), bf16 operands pre-converted so inner loops
//    contain no format conversion. __launch_bounds__(32,1) gives the
//    allocator the full VGPR file (no accumulator spills).
//  - conv weight tiles and rr-GEMM A tiles staged by the Tensor Data Mover
//    (tensor_load_to_lds + s_wait_tensorcnt); B staging overlaps the DMA.
// ---------------------------------------------------------------------------

typedef __attribute__((ext_vector_type(16))) __bf16        v16bf;
typedef __attribute__((ext_vector_type(8)))  float         v8f;
typedef __attribute__((ext_vector_type(4)))  unsigned int  v4u;
typedef __attribute__((ext_vector_type(8)))  unsigned int  v8u;

union FragAB { v16bf v; uint4 q[2]; };

#define LDS_STRIDE 40   // halves per row (80 B: 16B-aligned, bank-skewed)
#define USE_TDM 1

__device__ __forceinline__ unsigned short f2bf(float f) {
    unsigned int u = __float_as_uint(f);
    u += 0x7FFFu + ((u >> 16) & 1u);           // round-to-nearest-even
    return (unsigned short)(u >> 16);
}
__device__ __forceinline__ float sigmoidf(float x) { return 1.0f / (1.0f + __expf(-x)); }

// A-fragment (16x32 bf16): lane<16 -> row=lane, K=0..7 & 16..23
//                          lane>=16 -> row=lane-16, K=8..15 & 24..31
__device__ __forceinline__ v16bf load_frag_row(const unsigned short* lds, int lane) {
    int r = lane & 15, h = lane >> 4;
    FragAB u;
    u.q[0] = *(const uint4*)(lds + r * LDS_STRIDE + h * 8);
    u.q[1] = *(const uint4*)(lds + r * LDS_STRIDE + 16 + h * 8);
    return u.v;
}
// B-fragment (32x16 bf16) staged transposed [n][k]; lane: n=lane&15,
// K=(lane>>4)*8..+7 & 16+(lane>>4)*8..+7 (same order as A).
__device__ __forceinline__ v16bf load_frag_col(const unsigned short* lds, int lane) {
    int n = lane & 15, h = lane >> 4;
    FragAB u;
    u.q[0] = *(const uint4*)(lds + n * LDS_STRIDE + h * 8);
    u.q[1] = *(const uint4*)(lds + n * LDS_STRIDE + 16 + h * 8);
    return u.v;
}

// 16 contiguous bf16 global -> contiguous LDS (both 16B aligned)
__device__ __forceinline__ void stage_copy16(const unsigned short* __restrict__ src,
                                             unsigned short* dst) {
    uint4 a = ((const uint4*)src)[0];
    uint4 b = ((const uint4*)src)[1];
    ((uint4*)dst)[0] = a;
    ((uint4*)dst)[1] = b;
}
// 16 contiguous bf16 global -> transposed scatter into LDS column kk
__device__ __forceinline__ void stage_scatter16(const unsigned short* __restrict__ src,
                                                unsigned short* lds, int kk) {
    union { uint4 q[2]; unsigned short h[16]; } t;
    t.q[0] = ((const uint4*)src)[0];
    t.q[1] = ((const uint4*)src)[1];
    #pragma unroll
    for (int e = 0; e < 16; ++e) lds[e * LDS_STRIDE + kk] = t.h[e];
}

#if USE_TDM
// TDM load of a (rows x 32-half) tile (row stride ldA halves) into LDS with
// 16B padding every 64B so rows land at the 80B fragment stride.
__device__ __forceinline__ void tdm_load_tile_a(const unsigned short* gsrc,
                                                unsigned short* ldsA,
                                                unsigned K, unsigned long long ldA,
                                                unsigned rows) {
    unsigned long long ga = (unsigned long long)(size_t)gsrc;
    unsigned lds_addr = (unsigned)(size_t)ldsA;
    v4u g0;
    g0[0] = 1u;                                            // count=1 (valid)
    g0[1] = lds_addr;                                      // lds_addr
    g0[2] = (unsigned)ga;                                  // global_addr[31:0]
    g0[3] = ((unsigned)(ga >> 32) & 0x01FFFFFFu) | 0x80000000u; // addr[56:32]|type=2
    v8u g1;
    g1[0] = (1u << 16)        // data_size = 2 bytes
          | (1u << 20)        // pad_enable
          | (3u << 22)        // pad_interval: 16 dwords (64B)
          | (3u << 25);       // pad_amount: 4 dwords (16B) -> 80B rows
    g1[1] = (K & 0xFFFFu) << 16;                           // tensor_dim0 lo
    g1[2] = (K >> 16) | (rows << 16);                      // dim0 hi | tensor_dim1 lo
    g1[3] = (32u << 16);                                   // tile_dim0 = 32
    g1[4] = rows;                                          // tile_dim1 = rows
    g1[5] = (unsigned)ldA;                                 // tensor_dim0_stride lo
    g1[6] = (unsigned)((ldA >> 32) & 0xFFFFu);             // stride hi
    g1[7] = 0u;
    asm volatile("tensor_load_to_lds %0, %1" :: "s"(g0), "s"(g1) : "memory");
}
#endif

// ---------------------------------------------------------------------------
// 3x3 conv as implicit GEMM, bf16 in / bf16 w. Wave tile: 32(co) x 64(px).
// Weight tile staged by TDM; im2col B gather overlaps the DMA.
// out_mode: 0 = f32, 1 = bf16.
// ---------------------------------------------------------------------------
__global__ void __launch_bounds__(32, 1)
conv3x3_wmma(const unsigned short* __restrict__ in,
             const unsigned short* __restrict__ w,
             const float* __restrict__ bias,
             const float* __restrict__ addsrc,
             void* __restrict__ out, int out_mode,
             int Cin, int relu_in,
             long in_img_stride, int t_per,
             long out_b_stride, long out_t_stride, long out_c_stride) {
    __shared__ unsigned short ldsA[32 * LDS_STRIDE];
    __shared__ unsigned short ldsB[4][16 * LDS_STRIDE];

    const int lane = threadIdx.x;
    const int p0   = blockIdx.x * 64;
    const int co0  = blockIdx.y * 32;
    const int img  = blockIdx.z;
    const int b = img / t_per, t = img % t_per;

    const unsigned short* in_img = in + (long)img * in_img_stride;
    const long out_off = (long)b * out_b_stride + (long)t * out_t_stride;
    const int K = Cin * 9;

    // incremental im2col indices for k = k0 + lane (k0 += 32 => j += 5 mod 9)
    int ci = lane / 9;
    int j  = lane - ci * 9;

    v8f acc[2][4] = {};
    for (int k0 = 0; k0 < K; k0 += 32) {
#if USE_TDM
        tdm_load_tile_a(w + (long)co0 * K + k0, ldsA, (unsigned)K,
                        (unsigned long long)K, 32u);
#else
        {
            int r = lane & 15, h = lane >> 4;
            stage_copy16(w + (long)(co0 + r) * K + k0 + h * 16,
                         ldsA + r * LDS_STRIDE + h * 16);
            stage_copy16(w + (long)(co0 + 16 + r) * K + k0 + h * 16,
                         ldsA + (16 + r) * LDS_STRIDE + h * 16);
        }
#endif
        // B: im2col gather, transposed [n][k]
        {
            int ky = j / 3, kx = j - ky * 3;
            const unsigned short* srcc = in_img + (long)ci * 1024;
            #pragma unroll
            for (int i = 0; i < 4; ++i) {
                int pi = p0 + 16 * i;
                int y = (pi >> 5) + ky - 1;
                int xb0 = (pi & 31) + kx - 1;
                const unsigned short* src = srcc + y * 32;
                bool yok = (y >= 0) && (y < 32);
                #pragma unroll
                for (int e = 0; e < 16; ++e) {
                    int x = xb0 + e;
                    unsigned short v = (yok && x >= 0 && x < 32) ? src[x] : (unsigned short)0;
                    if (relu_in && (v & 0x8000u)) v = 0;   // bf16 relu
                    ldsB[i][e * LDS_STRIDE + lane] = v;
                }
            }
        }
#if USE_TDM
        __builtin_amdgcn_s_wait_tensorcnt(0);
#endif
        __syncthreads();
        v16bf a0 = load_frag_row(ldsA, lane);
        v16bf a1 = load_frag_row(ldsA + 16 * LDS_STRIDE, lane);
        #pragma unroll
        for (int i = 0; i < 4; ++i) {
            v16bf bf = load_frag_col(ldsB[i], lane);
            acc[0][i] = __builtin_amdgcn_wmma_f32_16x16x32_bf16(false, a0, false, bf, (short)0, acc[0][i], false, false);
            acc[1][i] = __builtin_amdgcn_wmma_f32_16x16x32_bf16(false, a1, false, bf, (short)0, acc[1][i], false, false);
        }
        __syncthreads();
        // advance im2col indices by 32 k (= 3*9 + 5)
        j += 5; ci += 3;
        if (j >= 9) { j -= 9; ++ci; }
    }

    const int n = lane & 15, h = lane >> 4;
    #pragma unroll
    for (int mi = 0; mi < 2; ++mi)
        #pragma unroll
        for (int i = 0; i < 4; ++i)
            #pragma unroll
            for (int v = 0; v < 8; ++v) {
                int m = mi * 16 + h * 8 + v;
                long idx = out_off + (long)(co0 + m) * out_c_stride + p0 + 16 * i + n;
                float rr = acc[mi][i][v];
                if (bias)   rr += bias[co0 + m];
                if (addsrc) rr += addsrc[idx];
                if (out_mode) ((unsigned short*)out)[idx] = f2bf(rr);
                else          ((float*)out)[idx] = rr;
            }
}

// ---------------------------------------------------------------------------
// Affinity: C[M,N] = alpha * A_T[K,M] * B[K,N], A/B bf16 K-major, C f32.
// Wave tile 32M x 64N.
// ---------------------------------------------------------------------------
__global__ void __launch_bounds__(32, 1)
gemm_tn_wmma(const unsigned short* __restrict__ A,
             const unsigned short* __restrict__ B,
             float* __restrict__ C,
             int K, long ldA, long ldB, long ldC, float alpha) {
    __shared__ unsigned short ldsA[32 * LDS_STRIDE];
    __shared__ unsigned short ldsB[4][16 * LDS_STRIDE];
    const int lane = threadIdx.x;
    const long m0 = (long)blockIdx.y * 32;
    const long n0 = (long)blockIdx.x * 64;

    v8f acc[2][4] = {};
    for (int k0 = 0; k0 < K; k0 += 32) {
        long k = k0 + lane;
        stage_scatter16(A + k * ldA + m0,      ldsA,                   lane);
        stage_scatter16(A + k * ldA + m0 + 16, ldsA + 16 * LDS_STRIDE, lane);
        #pragma unroll
        for (int i = 0; i < 4; ++i)
            stage_scatter16(B + k * ldB + n0 + 16 * i, ldsB[i], lane);
        __builtin_prefetch(A + (k + 32) * ldA + m0, 0, 3);
        __builtin_prefetch(B + (k + 32) * ldB + n0, 0, 3);
        __syncthreads();
        v16bf a0 = load_frag_row(ldsA, lane);
        v16bf a1 = load_frag_row(ldsA + 16 * LDS_STRIDE, lane);
        #pragma unroll
        for (int i = 0; i < 4; ++i) {
            v16bf bf = load_frag_col(ldsB[i], lane);
            acc[0][i] = __builtin_amdgcn_wmma_f32_16x16x32_bf16(false, a0, false, bf, (short)0, acc[0][i], false, false);
            acc[1][i] = __builtin_amdgcn_wmma_f32_16x16x32_bf16(false, a1, false, bf, (short)0, acc[1][i], false, false);
        }
        __syncthreads();
    }
    const int n = lane & 15, h = lane >> 4;
    #pragma unroll
    for (int mi = 0; mi < 2; ++mi)
        #pragma unroll
        for (int i = 0; i < 4; ++i)
            #pragma unroll
            for (int v = 0; v < 8; ++v) {
                int m = mi * 16 + h * 8 + v;
                C[(m0 + m) * ldC + n0 + 16 * i + n] = alpha * acc[mi][i][v];
            }
}

// ---------------------------------------------------------------------------
// C[M,N] = A[M,K] * B[K,N] + bias (A,B bf16 row-major, C f32, opt relu).
// 32-row A tile staged by the Tensor Data Mover; B staging overlaps the DMA.
// ---------------------------------------------------------------------------
__global__ void __launch_bounds__(32, 1)
gemm_rr_wmma(const unsigned short* __restrict__ A,
             const unsigned short* __restrict__ B,
             const float* __restrict__ bias,
             float* __restrict__ C,
             int K, long ldA, long ldB, long ldC, int relu_out) {
    __shared__ unsigned short ldsA[32 * LDS_STRIDE];
    __shared__ unsigned short ldsB[4][16 * LDS_STRIDE];
    const int lane = threadIdx.x;
    const long m0 = (long)blockIdx.y * 32;
    const long n0 = (long)blockIdx.x * 64;
    const int r = lane & 15, h = lane >> 4;

    v8f acc[2][4] = {};
    for (int k0 = 0; k0 < K; k0 += 32) {
#if USE_TDM
        tdm_load_tile_a(A + m0 * ldA + k0, ldsA, (unsigned)K,
                        (unsigned long long)ldA, 32u);
#else
        stage_copy16(A + (m0 + r) * ldA + k0 + h * 16, ldsA + r * LDS_STRIDE + h * 16);
        stage_copy16(A + (m0 + 16 + r) * ldA + k0 + h * 16, ldsA + (16 + r) * LDS_STRIDE + h * 16);
#endif
        long k = k0 + lane;
        #pragma unroll
        for (int i = 0; i < 4; ++i)
            stage_scatter16(B + k * ldB + n0 + 16 * i, ldsB[i], lane);
        __builtin_prefetch(B + (k + 32) * ldB + n0, 0, 3);
#if USE_TDM
        __builtin_amdgcn_s_wait_tensorcnt(0);
#endif
        __syncthreads();
        v16bf a0 = load_frag_row(ldsA, lane);
        v16bf a1 = load_frag_row(ldsA + 16 * LDS_STRIDE, lane);
        #pragma unroll
        for (int i = 0; i < 4; ++i) {
            v16bf bf = load_frag_col(ldsB[i], lane);
            acc[0][i] = __builtin_amdgcn_wmma_f32_16x16x32_bf16(false, a0, false, bf, (short)0, acc[0][i], false, false);
            acc[1][i] = __builtin_amdgcn_wmma_f32_16x16x32_bf16(false, a1, false, bf, (short)0, acc[1][i], false, false);
        }
        __syncthreads();
    }
    const int n = lane & 15;
    #pragma unroll
    for (int mi = 0; mi < 2; ++mi)
        #pragma unroll
        for (int i = 0; i < 4; ++i)
            #pragma unroll
            for (int v = 0; v < 8; ++v) {
                int m = mi * 16 + h * 8 + v;
                float rr = acc[mi][i][v];
                if (bias) rr += bias[m0 + m];
                if (relu_out) rr = fmaxf(rr, 0.0f);
                C[(m0 + m) * ldC + n0 + 16 * i + n] = rr;
            }
}

// ---------------------------------------------------------------------------
// f32 -> bf16 elementwise conversion (one-time prep)
// ---------------------------------------------------------------------------
__global__ void cvt_f32_bf16_kernel(const float* __restrict__ src,
                                    unsigned short* __restrict__ dst, int n) {
    int i = blockIdx.x * blockDim.x + threadIdx.x;
    if (i < n) dst[i] = f2bf(src[i]);
}

// ---------------------------------------------------------------------------
// Column softmax over M rows, bf16 output. Block 16x16 (16 cols).
// ---------------------------------------------------------------------------
__global__ void softmax_col_bf16(const float* __restrict__ aff,
                                 unsigned short* __restrict__ affb, int M, int N) {
    __shared__ float red[16][17];
    const int tx = threadIdx.x, ty = threadIdx.y;
    const int n = blockIdx.x * 16 + tx;

    float mx = -3.0e38f;
    for (int m = ty; m < M; m += 16) mx = fmaxf(mx, aff[(long)m * N + n]);
    red[ty][tx] = mx; __syncthreads();
    if (ty == 0) {
        float v = red[0][tx];
        #pragma unroll
        for (int i = 1; i < 16; ++i) v = fmaxf(v, red[i][tx]);
        red[0][tx] = v;
    }
    __syncthreads();
    mx = red[0][tx];
    __syncthreads();

    float sm = 0.0f;
    for (int m = ty; m < M; m += 16) sm += __expf(aff[(long)m * N + n] - mx);
    red[ty][tx] = sm; __syncthreads();
    if (ty == 0) {
        float v = 0.0f;
        #pragma unroll
        for (int i = 0; i < 16; ++i) v += red[i][tx];
        red[0][tx] = v;
    }
    __syncthreads();
    float inv = 1.0f / red[0][tx];
    for (int m = ty; m < M; m += 16)
        affb[(long)m * N + n] = f2bf(__expf(aff[(long)m * N + n] - mx) * inv);
}

// x = concat([f16_q, val]) -> bf16 [B, 512, 1024]
__global__ void concat_x_kernel(const float* __restrict__ f16_q,
                                const float* __restrict__ val,
                                unsigned short* __restrict__ x, int total) {
    int i = blockIdx.x * blockDim.x + threadIdx.x;
    if (i >= total) return;
    int p = i & 1023, c = (i >> 10) & 511, b = i >> 19;
    float v = (c < 256) ? f16_q[(long)b * 262144 + c * 1024 + p]
                        : val[(long)b * 262144 + (c - 256) * 1024 + p];
    x[i] = f2bf(v);
}

// ---------------------------------------------------------------------------
// CBAM
// ---------------------------------------------------------------------------
__global__ void cbam_channel_kernel(const float* __restrict__ xb,
                                    const float* __restrict__ w1, const float* __restrict__ b1,
                                    const float* __restrict__ w2, const float* __restrict__ b2,
                                    float* __restrict__ ch) {
    __shared__ float smean[256], smax[256], hm[16], hx[16];
    const int b = blockIdx.x, c = threadIdx.x;
    const float* base = xb + (long)b * 262144 + (long)c * 1024;
    float s = 0.0f, m = -3.0e38f;
    for (int p = 0; p < 1024; ++p) { float v = base[p]; s += v; m = fmaxf(m, v); }
    smean[c] = s * (1.0f / 1024.0f);
    smax[c] = m;
    __syncthreads();
    if (c < 16) {
        float am = b1[c], ax = b1[c];
        const float* wr = w1 + c * 256;
        for (int i = 0; i < 256; ++i) { am += wr[i] * smean[i]; ax += wr[i] * smax[i]; }
        hm[c] = fmaxf(am, 0.0f);
        hx[c] = fmaxf(ax, 0.0f);
    }
    __syncthreads();
    float o = 2.0f * b2[c];
    const float* w2r = w2 + c * 16;
    #pragma unroll
    for (int j = 0; j < 16; ++j) o += (hm[j] + hx[j]) * w2r[j];
    ch[b * 256 + c] = sigmoidf(o);
}

__global__ void cbam_comp_kernel(const float* __restrict__ xb, const float* __restrict__ ch,
                                 float* __restrict__ comp, int total) {
    int i = blockIdx.x * blockDim.x + threadIdx.x;
    if (i >= total) return;
    int p = i & 1023, b = i >> 10;
    float mx = -3.0e38f, sm = 0.0f;
    const float* xbb = xb + (long)b * 262144 + p;
    const float* chb = ch + b * 256;
    for (int c = 0; c < 256; ++c) {
        float v = xbb[c * 1024] * chb[c];
        mx = fmaxf(mx, v); sm += v;
    }
    comp[(long)b * 2048 + p] = mx;
    comp[(long)b * 2048 + 1024 + p] = sm * (1.0f / 256.0f);
}

__global__ void cbam_sgate_kernel(const float* __restrict__ comp,
                                  const float* __restrict__ sp_w,
                                  const float* __restrict__ sp_g,
                                  const float* __restrict__ sp_b,
                                  float* __restrict__ sg, int total) {
    int i = blockIdx.x * blockDim.x + threadIdx.x;
    if (i >= total) return;
    int p = i & 1023, b = i >> 10;
    int y = p >> 5, x = p & 31;
    float acc = 0.0f;
    for (int ic = 0; ic < 2; ++ic)
        for (int dy = 0; dy < 7; ++dy) {
            int yy = y + dy - 3;
            if (yy < 0 || yy >= 32) continue;
            for (int dx = 0; dx < 7; ++dx) {
                int xx = x + dx - 3;
                if (xx < 0 || xx >= 32) continue;
                acc += sp_w[ic * 49 + dy * 7 + dx] * comp[(long)b * 2048 + ic * 1024 + yy * 32 + xx];
            }
        }
    sg[i] = acc * (sp_g[0] * rsqrtf(1.0f + 1e-5f)) + sp_b[0];
}

// fused(f32) = xb*(1+ch*sig(sg)); also bf16 into bott_in channels 256..511
__global__ void cbam_fuse_kernel(const float* __restrict__ xb, const float* __restrict__ ch,
                                 const float* __restrict__ sg,
                                 float* __restrict__ fused, unsigned short* __restrict__ bott_in,
                                 int total) {
    int i = blockIdx.x * blockDim.x + threadIdx.x;
    if (i >= total) return;
    int p = i & 1023, c = (i >> 10) & 255, b = i >> 18;
    float g = 1.0f + ch[b * 256 + c] * sigmoidf(sg[b * 1024 + p]);
    float v = xb[i] * g;
    fused[i] = v;
    bott_in[(long)b * 524288 + (long)(256 + c) * 1024 + p] = f2bf(v);
}

// ---------------------------------------------------------------------------
// PSP pyramid
// ---------------------------------------------------------------------------
__global__ void psp_pool_kernel(const float* __restrict__ fused, float* __restrict__ pooled,
                                int s, int total) {
    int i = blockIdx.x * blockDim.x + threadIdx.x;
    if (i >= total) return;
    int ss = s * s;
    int cell = i % ss, c = (i / ss) & 255, b = i / (ss * 256);
    int ci = cell / s, cj = cell % s, blk = 32 / s;
    const float* base = fused + (long)b * 262144 + (long)c * 1024 + (ci * blk) * 32 + cj * blk;
    float acc = 0.0f;
    for (int y = 0; y < blk; ++y)
        for (int x = 0; x < blk; ++x) acc += base[y * 32 + x];
    pooled[i] = acc / (float)(blk * blk);
}

__global__ void psp_conv_kernel(const float* __restrict__ pooled, const float* __restrict__ w,
                                float* __restrict__ pc, int s, int total) {
    int i = blockIdx.x * blockDim.x + threadIdx.x;
    if (i >= total) return;
    int ss = s * s;
    int cell = i % ss, co = (i / ss) & 63, b = i / (ss * 64);
    const float* pb = pooled + (long)b * 256 * ss + cell;
    const float* wr = w + co * 256;
    float acc = 0.0f;
    for (int c = 0; c < 256; ++c) acc += wr[c] * pb[c * ss];
    pc[i] = acc;
}

__device__ __forceinline__ void interpw(int i, int s, int& lo, int& hi, float& w) {
    if (s == 1) { lo = 0; hi = 0; w = 0.0f; return; }
    float c = (float)i * (float)(s - 1) / 31.0f;
    int l = (int)floorf(c);
    if (l > s - 1) l = s - 1;
    lo = l; hi = min(l + 1, s - 1); w = c - (float)l;
}

__global__ void psp_upsample_kernel(const float* __restrict__ pc,
                                    unsigned short* __restrict__ bott_in,
                                    int s, int choff, int total) {
    int i = blockIdx.x * blockDim.x + threadIdx.x;
    if (i >= total) return;
    int p = i & 1023, cc = (i >> 10) & 63, b = i >> 16;
    int yi = p >> 5, xi = p & 31;
    int rlo, rhi, clo, chi; float rw, cw;
    interpw(yi, s, rlo, rhi, rw);
    interpw(xi, s, clo, chi, cw);
    const float* P = pc + ((long)b * 64 + cc) * s * s;
    float v = (1.0f - rw) * ((1.0f - cw) * P[rlo * s + clo] + cw * P[rlo * s + chi])
            + rw          * ((1.0f - cw) * P[rhi * s + clo] + cw * P[rhi * s + chi]);
    bott_in[(long)b * 524288 + (long)(choff + cc) * 1024 + p] = f2bf(v);
}

// ---------------------------------------------------------------------------
// Host side
// ---------------------------------------------------------------------------
static inline int ceil_div(int a, int b) { return (a + b - 1) / b; }

extern "C" void kernel_launch(void* const* d_in, const int* in_sizes, int n_in,
                              void* d_out, int out_size, void* d_ws, size_t ws_size,
                              hipStream_t stream) {
    const float* f16_q  = (const float*)d_in[0];
    const float* f16_m  = (const float*)d_in[1];
    const float* value_m= (const float*)d_in[2];
    const float* pk_w   = (const float*)d_in[3];
    const float* pk_b   = (const float*)d_in[4];
    const float* rb1_w  = (const float*)d_in[5];
    const float* rb1_b  = (const float*)d_in[6];
    const float* rb2_w  = (const float*)d_in[7];
    const float* rb2_b  = (const float*)d_in[8];
    const float* rbd_w  = (const float*)d_in[9];
    const float* rbd_b  = (const float*)d_in[10];
    const float* mlp_w1 = (const float*)d_in[11];
    const float* mlp_b1 = (const float*)d_in[12];
    const float* mlp_w2 = (const float*)d_in[13];
    const float* mlp_b2 = (const float*)d_in[14];
    const float* sp_w   = (const float*)d_in[15];
    const float* sp_g   = (const float*)d_in[16];
    const float* sp_b   = (const float*)d_in[17];
    const float* psp_w[4] = { (const float*)d_in[18], (const float*)d_in[19],
                              (const float*)d_in[20], (const float*)d_in[21] };
    const float* bott_w = (const float*)d_in[22];
    const float* bott_b = (const float*)d_in[23];
    float* out = (float*)d_out;

    typedef unsigned short u16;
    char* ws = (char*)d_ws;
    size_t off = 0;
    auto alloc = [&](size_t bytes) -> char* {
        char* p = ws + off;
        off = (off + bytes + 255) & ~(size_t)255;
        return p;
    };
    // bf16 prepped operands
    u16* m_bf    = (u16*)alloc(8388608ul * 2);   // f16_m
    u16* q_bf    = (u16*)alloc(1048576ul * 2);   // f16_q
    u16* v_bf    = (u16*)alloc(8388608ul * 2);   // value_m
    u16* pkw_bf  = (u16*)alloc(147456ul * 2);
    u16* rb1w_bf = (u16*)alloc(1179648ul * 2);
    u16* rb2w_bf = (u16*)alloc(589824ul * 2);
    u16* rbdw_bf = (u16*)alloc(1179648ul * 2);
    u16* bottw_bf= (u16*)alloc(131072ul * 2);
    // activations
    u16*   mkb    = (u16*)alloc(2097152ul * 2);  // [B,64,8192] bf16
    u16*   qkb    = (u16*)alloc(262144ul * 2);   // [B,64,1024] bf16
    float* aff    = (float*)alloc(8388608ul * 4);// [8192,1024] per-batch reuse
    u16*   affb   = (u16*)alloc(8388608ul * 2);
    float* val    = (float*)alloc(1048576ul * 4);
    u16*   xcatb  = (u16*)alloc(2097152ul * 2);  // [B,512,1024] bf16
    u16*   r1b    = (u16*)alloc(1048576ul * 2);  // bf16
    float* r2     = (float*)alloc(1048576ul * 4);
    float* xbuf   = (float*)alloc(1048576ul * 4);
    float* ch     = (float*)alloc(1024ul * 4);
    float* comp   = (float*)alloc(8192ul * 4);
    float* sg     = (float*)alloc(4096ul * 4);
    float* fused  = (float*)alloc(1048576ul * 4);
    float* pooled = (float*)alloc(87040ul * 4);
    float* pconv  = (float*)alloc(21760ul * 4);
    u16*   bott_in= (u16*)alloc(2097152ul * 2);  // [B,512,1024] bf16

    // --- 0. one-time bf16 conversions ---
    struct { const float* s; u16* d; int n; } cv[] = {
        { f16_m, m_bf, 8388608 }, { f16_q, q_bf, 1048576 }, { value_m, v_bf, 8388608 },
        { pk_w, pkw_bf, 147456 }, { rb1_w, rb1w_bf, 1179648 }, { rb2_w, rb2w_bf, 589824 },
        { rbd_w, rbdw_bf, 1179648 }, { bott_w, bottw_bf, 131072 },
    };
    for (auto& c : cv)
        cvt_f32_bf16_kernel<<<ceil_div(c.n, 256), 256, 0, stream>>>(c.s, c.d, c.n);

    // --- 1. key encoder conv (bf16 out straight into mk/qk layouts) ---
    conv3x3_wmma<<<dim3(16, 2, 32), 32, 0, stream>>>(
        m_bf, pkw_bf, pk_b, nullptr, mkb, 1,
        256, 0, 262144l, 8, 524288l, 1024l, 8192l);
    conv3x3_wmma<<<dim3(16, 2, 4), 32, 0, stream>>>(
        q_bf, pkw_bf, pk_b, nullptr, qkb, 1,
        256, 0, 262144l, 1, 65536l, 1024l, 1024l);

    // --- 2. attention (batch-serialized, aff buffers reused) ---
    for (int b = 0; b < 4; ++b) {
        gemm_tn_wmma<<<dim3(16, 256), 32, 0, stream>>>(
            mkb + (long)b * 524288, qkb + (long)b * 65536, aff,
            64, 8192l, 1024l, 1024l, 0.125f);
        softmax_col_bf16<<<dim3(64), dim3(16, 16), 0, stream>>>(aff, affb, 8192, 1024);
        gemm_rr_wmma<<<dim3(16, 8), 32, 0, stream>>>(
            v_bf + (long)b * 2097152, affb, nullptr,
            val + (long)b * 262144, 8192, 8192l, 1024l, 1024l, 0);
    }

    // --- 3. concat -> bf16 x ---
    concat_x_kernel<<<ceil_div(4 * 512 * 1024, 256), 256, 0, stream>>>(
        f16_q, val, xcatb, 4 * 512 * 1024);

    // --- 4. ResBlock ---
    conv3x3_wmma<<<dim3(16, 8, 4), 32, 0, stream>>>(    // r1 = conv(relu(x)), bf16 out
        xcatb, rb1w_bf, rb1_b, nullptr, r1b, 1,
        512, 1, 524288l, 1, 262144l, 0l, 1024l);
    conv3x3_wmma<<<dim3(16, 8, 4), 32, 0, stream>>>(    // r2 = conv(relu(r1)), f32 out
        r1b, rb2w_bf, rb2_b, nullptr, r2, 0,
        256, 1, 262144l, 1, 262144l, 0l, 1024l);
    conv3x3_wmma<<<dim3(16, 8, 4), 32, 0, stream>>>(    // xb = conv(x) + r2
        xcatb, rbdw_bf, rbd_b, r2, xbuf, 0,
        512, 0, 524288l, 1, 262144l, 0l, 1024l);

    // --- 5. CBAM ---
    cbam_channel_kernel<<<4, 256, 0, stream>>>(xbuf, mlp_w1, mlp_b1, mlp_w2, mlp_b2, ch);
    cbam_comp_kernel<<<ceil_div(4096, 256), 256, 0, stream>>>(xbuf, ch, comp, 4096);
    cbam_sgate_kernel<<<ceil_div(4096, 256), 256, 0, stream>>>(comp, sp_w, sp_g, sp_b, sg, 4096);
    cbam_fuse_kernel<<<ceil_div(1048576, 256), 256, 0, stream>>>(xbuf, ch, sg, fused, bott_in, 1048576);

    // --- 6. PSP -> bott_in channels 0..255 (bf16) ---
    {
        const int sizes[4] = {1, 2, 4, 8};
        const int cumcell[4] = {0, 1, 5, 21};
        const int choff[4] = {0, 64, 128, 192};
        for (int i = 0; i < 4; ++i) {
            int s = sizes[i], ss = s * s;
            float* pool_s = pooled + (long)cumcell[i] * 4 * 256;
            float* pc_s   = pconv  + (long)cumcell[i] * 4 * 64;
            psp_pool_kernel<<<ceil_div(4 * 256 * ss, 256), 256, 0, stream>>>(fused, pool_s, s, 4 * 256 * ss);
            psp_conv_kernel<<<ceil_div(4 * 64 * ss, 256), 256, 0, stream>>>(pool_s, psp_w[i], pc_s, s, 4 * 64 * ss);
            psp_upsample_kernel<<<ceil_div(4 * 64 * 1024, 256), 256, 0, stream>>>(pc_s, bott_in, s, choff[i], 4 * 64 * 1024);
        }
    }

    // --- 7. bottleneck 1x1 + ReLU -> d_out ---
    for (int b = 0; b < 4; ++b) {
        gemm_rr_wmma<<<dim3(16, 8), 32, 0, stream>>>(
            bottw_bf, bott_in + (long)b * 524288, bott_b,
            out + (long)b * 262144, 512, 512l, 1024l, 1024l, 1);
    }

    (void)in_sizes; (void)n_in; (void)out_size; (void)ws_size;
}
